// Set2SetPool_1967095021850
// MI455X (gfx1250) — matrix operand order repaired and checked by reading, not compiled, not run on hardware
//
#include <hip/hip_runtime.h>
#include <hip/hip_bf16.h>
#include <math.h>

// ---------------------------------------------------------------------------
// Set2Set pooling for MI455X (gfx1250, wave32, WMMA).
// GEMMs use v_wmma_f32_16x16x32_bf16 (f32 accumulate); attention/segment ops
// are bandwidth-bound fp32 VALU kernels.
// ---------------------------------------------------------------------------

typedef __attribute__((ext_vector_type(16))) __bf16 v16bf;
typedef __attribute__((ext_vector_type(8)))  __bf16 v8bf;
typedef __attribute__((ext_vector_type(8)))  float  v8f;

#define DIN      512
#define B_GRAPHS 1024
#define NNODES   100000
#define KCOMB    1536          // [q(512) | r(512) | h(512)] -> q_star(1024)+h(512)
#define NGATES   2048          // 4*DIN
#define DOUT_    512

__device__ __forceinline__ float sigmoidf_(float v) { return 1.0f / (1.0f + expf(-v)); }

// ---------------------------------------------------------------------------
// GEMM: C[M,N] = act( A[M,K](bf16, stride lda) * W[N,K](bf16, stride ldw)^T + bias )
// One wave computes a 32x32 tile with 2x2 WMMA accumulators.
// A fragment (16x32, 16-bit): lane L(0-15) row M=L, v16 elems 0..15 = K hi*16..hi*16+15
//   (hi = lane>>4)  -> two 16B loads.
// B fragment (32x16, 16-bit): lane L column N=L&15, 16 contiguous K starting at
//   hi*16 -> one 32B load of a W row (A.B^T pattern, both row-major).
// D (16x16 f32): lane L col N=L&15, VGPR v row M=hi*8+v.
// ---------------------------------------------------------------------------
__global__ void gemm_bf16_wmma(const __bf16* __restrict__ A, int lda,
                               const __bf16* __restrict__ W, int ldw,
                               const float* __restrict__ bias,
                               float* __restrict__ C, int ldc,
                               int M, int N, int K, int relu)
{
    const int lane  = threadIdx.x & 31;
    const int wave  = blockIdx.x * (blockDim.x >> 5) + (threadIdx.x >> 5);
    const int tilesN = N >> 5;
    const int tm = wave / tilesN;
    const int tn = wave - tm * tilesN;
    if ((tm << 5) >= M) return;                 // uniform per-wave; EXEC stays all-1

    const int l15 = lane & 15;
    const int hi  = lane >> 4;

    const __bf16* a0p = A + (size_t)(tm * 32 + l15) * lda + hi * 16;
    const __bf16* a1p = a0p + (size_t)16 * lda;
    const __bf16* b0p = W + (size_t)(tn * 32 + l15) * ldw + hi * 16;
    const __bf16* b1p = b0p + (size_t)16 * ldw;

    v8f acc00 = {}, acc01 = {}, acc10 = {}, acc11 = {};

    for (int k = 0; k < K; k += 32) {
        v8bf a0lo = *(const v8bf*)(a0p + k);
        v8bf a0hi = *(const v8bf*)(a0p + k + 8);
        v8bf a1lo = *(const v8bf*)(a1p + k);
        v8bf a1hi = *(const v8bf*)(a1p + k + 8);
        v16bf a0 = __builtin_shufflevector(a0lo, a0hi, 0,1,2,3,4,5,6,7,8,9,10,11,12,13,14,15);
        v16bf a1 = __builtin_shufflevector(a1lo, a1hi, 0,1,2,3,4,5,6,7,8,9,10,11,12,13,14,15);
        v16bf b0 = *(const v16bf*)(b0p + k);
        v16bf b1 = *(const v16bf*)(b1p + k);
        acc00 = __builtin_amdgcn_wmma_f32_16x16x32_bf16(false, a0, false, b0, (short)0, acc00, false, false);
        acc01 = __builtin_amdgcn_wmma_f32_16x16x32_bf16(false, a0, false, b1, (short)0, acc01, false, false);
        acc10 = __builtin_amdgcn_wmma_f32_16x16x32_bf16(false, a1, false, b0, (short)0, acc10, false, false);
        acc11 = __builtin_amdgcn_wmma_f32_16x16x32_bf16(false, a1, false, b1, (short)0, acc11, false, false);
    }

    const int nc0 = tn * 32 + l15;
    const int nc1 = nc0 + 16;
    const float bv0 = bias ? bias[nc0] : 0.0f;
    const float bv1 = bias ? bias[nc1] : 0.0f;
    const int mb = tm * 32 + hi * 8;
#pragma unroll
    for (int v = 0; v < 8; ++v) {
        float r00 = acc00[v] + bv0;
        float r01 = acc01[v] + bv1;
        float r10 = acc10[v] + bv0;
        float r11 = acc11[v] + bv1;
        if (relu) {
            r00 = fmaxf(r00, 0.0f); r01 = fmaxf(r01, 0.0f);
            r10 = fmaxf(r10, 0.0f); r11 = fmaxf(r11, 0.0f);
        }
        C[(size_t)(mb + v) * ldc + nc0]      = r00;
        C[(size_t)(mb + v) * ldc + nc1]      = r01;
        C[(size_t)(mb + 16 + v) * ldc + nc0] = r10;
        C[(size_t)(mb + 16 + v) * ldc + nc1] = r11;
    }
}

// --------------------------- prep kernels ----------------------------------
__global__ void prep_wcomb(const float* __restrict__ Wih, const float* __restrict__ Whh,
                           __bf16* __restrict__ Wcomb)
{
    int idx = blockIdx.x * blockDim.x + threadIdx.x;
    if (idx >= NGATES * KCOMB) return;
    int r = idx / KCOMB, cc = idx - r * KCOMB;
    float v = (cc < 2 * DIN) ? Wih[(size_t)r * (2 * DIN) + cc]
                             : Whh[(size_t)r * DIN + (cc - 2 * DIN)];
    Wcomb[idx] = (__bf16)v;
}

__global__ void prep_wpost(const float* __restrict__ Wpost, __bf16* __restrict__ Wp)
{
    int idx = blockIdx.x * blockDim.x + threadIdx.x;
    if (idx >= DOUT_ * 2 * DIN) return;
    Wp[idx] = (__bf16)Wpost[idx];
}

__global__ void prep_bias(const float* __restrict__ bih, const float* __restrict__ bhh,
                          float* __restrict__ bias)
{
    int idx = blockIdx.x * blockDim.x + threadIdx.x;
    if (idx >= NGATES) return;
    bias[idx] = bih[idx] + bhh[idx];
}

// CSR row starts from sorted batch (handles empty graphs).
__global__ void build_rowstart(const int* __restrict__ batch, int* __restrict__ rs)
{
    int i = blockIdx.x * blockDim.x + threadIdx.x;
    if (i >= NNODES) return;
    int b  = batch[i];
    int bp = (i == 0) ? -1 : batch[i - 1];
    for (int g = bp + 1; g <= b; ++g) rs[g] = i;
    if (i == NNODES - 1)
        for (int g = b + 1; g <= B_GRAPHS; ++g) rs[g] = NNODES;
}

__global__ void init_state(__bf16* __restrict__ inbf, float* __restrict__ h, float* __restrict__ c)
{
    int idx = blockIdx.x * blockDim.x + threadIdx.x;
    if (idx >= B_GRAPHS * KCOMB) return;
    inbf[idx] = (__bf16)0.0f;
    if (idx < B_GRAPHS * DIN) { h[idx] = 0.0f; c[idx] = 0.0f; }
}

// --------------------------- LSTM gates ------------------------------------
__global__ void lstm_gate(const float* __restrict__ z, float* __restrict__ c,
                          float* __restrict__ h, __bf16* __restrict__ inbf)
{
    int idx = blockIdx.x * blockDim.x + threadIdx.x;
    if (idx >= B_GRAPHS * DIN) return;
    int b = idx >> 9;            // / DIN
    int d = idx & (DIN - 1);
    const float* zr = z + (size_t)b * NGATES;
    float ig = sigmoidf_(zr[d]);
    float fg = sigmoidf_(zr[DIN + d]);
    float gg = tanhf(zr[2 * DIN + d]);
    float og = sigmoidf_(zr[3 * DIN + d]);
    float cn = fg * c[idx] + ig * gg;
    float hn = og * tanhf(cn);
    c[idx] = cn;
    h[idx] = hn;
    __bf16 hb = (__bf16)hn;
    inbf[(size_t)b * KCOMB + d]            = hb;   // q part of q_star
    inbf[(size_t)b * KCOMB + 2 * DIN + d]  = hb;   // h input
}

// --------------------------- attention -------------------------------------
// wave per node: e[i] = dot(x[i,:], h[batch[i],:])
__global__ void attn_e(const float* __restrict__ x, const int* __restrict__ batch,
                       const float* __restrict__ h, float* __restrict__ e)
{
    int gid  = blockIdx.x * blockDim.x + threadIdx.x;
    int node = gid >> 5;
    int lane = gid & 31;
    if (node >= NNODES) return;
    int b = batch[node];
    const float4* xr = (const float4*)(x + (size_t)node * DIN);
    const float4* qr = (const float4*)(h + (size_t)b * DIN);
    float s = 0.0f;
#pragma unroll
    for (int j = lane; j < DIN / 4; j += 32) {
        float4 xv = xr[j], qv = qr[j];
        s = fmaf(xv.x, qv.x, s); s = fmaf(xv.y, qv.y, s);
        s = fmaf(xv.z, qv.z, s); s = fmaf(xv.w, qv.w, s);
    }
#pragma unroll
    for (int off = 16; off > 0; off >>= 1) s += __shfl_xor(s, off, 32);
    if (lane == 0) e[node] = s;
}

// block per graph: segment max + sum(exp(e-max))
__global__ void seg_reduce(const float* __restrict__ e, const int* __restrict__ rs,
                           float* __restrict__ emax, float* __restrict__ denom)
{
    __shared__ float sm[256];
    int g = blockIdx.x, tid = threadIdx.x;
    int s0 = rs[g], s1 = rs[g + 1];
    float m = -INFINITY;
    for (int i = s0 + tid; i < s1; i += 256) m = fmaxf(m, e[i]);
    sm[tid] = m; __syncthreads();
    for (int w = 128; w > 0; w >>= 1) { if (tid < w) sm[tid] = fmaxf(sm[tid], sm[tid + w]); __syncthreads(); }
    m = sm[0]; __syncthreads();
    float s = 0.0f;
    for (int i = s0 + tid; i < s1; i += 256) s += expf(e[i] - m);
    sm[tid] = s; __syncthreads();
    for (int w = 128; w > 0; w >>= 1) { if (tid < w) sm[tid] += sm[tid + w]; __syncthreads(); }
    if (tid == 0) { emax[g] = m; denom[g] = sm[0]; }
}

__global__ void attn_coef(const float* __restrict__ e, const int* __restrict__ batch,
                          const float* __restrict__ emax, const float* __restrict__ denom,
                          float* __restrict__ a)
{
    int i = blockIdx.x * blockDim.x + threadIdx.x;
    if (i >= NNODES) return;
    int b = batch[i];
    float d = denom[b];
    a[i] = (d > 0.0f) ? expf(e[i] - emax[b]) / d : 0.0f;
}

// block(256) per graph; each thread owns 2 columns: r[g,:] = sum a[i]*x[i,:]
__global__ void seg_wsum(const float* __restrict__ x, const float* __restrict__ a,
                         const int* __restrict__ rs, __bf16* __restrict__ inbf)
{
    int g = blockIdx.x;
    int s0 = rs[g], s1 = rs[g + 1];
    int c0 = threadIdx.x * 2;
    float r0 = 0.0f, r1 = 0.0f;
    for (int i = s0; i < s1; ++i) {
        float ai = a[i];
        float2 xv = *(const float2*)(x + (size_t)i * DIN + c0);
        r0 = fmaf(ai, xv.x, r0);
        r1 = fmaf(ai, xv.y, r1);
    }
    size_t ib = (size_t)g * KCOMB + DIN + c0;   // r part of q_star
    inbf[ib]     = (__bf16)r0;
    inbf[ib + 1] = (__bf16)r1;
}

// ---------------------------------------------------------------------------
extern "C" void kernel_launch(void* const* d_in, const int* in_sizes, int n_in,
                              void* d_out, int out_size, void* d_ws, size_t ws_size,
                              hipStream_t stream)
{
    const float* x     = (const float*)d_in[0];
    const int*   batch = (const int*)d_in[1];
    const float* Wih   = (const float*)d_in[2];
    const float* Whh   = (const float*)d_in[3];
    const float* bih   = (const float*)d_in[4];
    const float* bhh   = (const float*)d_in[5];
    const float* Wpost = (const float*)d_in[6];
    const float* bpost = (const float*)d_in[7];
    float* out = (float*)d_out;

    // workspace layout (~24 MB), 256B aligned slices
    char* ws = (char*)d_ws;
    size_t off = 0;
    auto alloc = [&](size_t bytes) -> char* {
        char* p = ws + off;
        off = (off + bytes + 255) & ~(size_t)255;
        return p;
    };
    __bf16* Wcomb = (__bf16*)alloc((size_t)NGATES * KCOMB * 2);
    __bf16* Wp    = (__bf16*)alloc((size_t)DOUT_ * 2 * DIN * 2);
    float*  biasc = (float*) alloc((size_t)NGATES * 4);
    __bf16* inbf  = (__bf16*)alloc((size_t)B_GRAPHS * KCOMB * 2);
    float*  z     = (float*) alloc((size_t)B_GRAPHS * NGATES * 4);
    float*  h     = (float*) alloc((size_t)B_GRAPHS * DIN * 4);
    float*  c     = (float*) alloc((size_t)B_GRAPHS * DIN * 4);
    float*  e     = (float*) alloc((size_t)NNODES * 4);
    float*  ac    = (float*) alloc((size_t)NNODES * 4);
    float*  emax  = (float*) alloc((size_t)B_GRAPHS * 4);
    float*  denom = (float*) alloc((size_t)B_GRAPHS * 4);
    int*    rs    = (int*)   alloc((size_t)(B_GRAPHS + 1) * 4);
    (void)ws_size; (void)in_sizes; (void)n_in; (void)out_size; (void)bpost;

    const int T = 256;
    // prep (recomputed every launch: deterministic, cheap vs. main work)
    prep_wcomb   <<<(NGATES * KCOMB + T - 1) / T, T, 0, stream>>>(Wih, Whh, Wcomb);
    prep_wpost   <<<(DOUT_ * 2 * DIN + T - 1) / T, T, 0, stream>>>(Wpost, Wp);
    prep_bias    <<<(NGATES + T - 1) / T, T, 0, stream>>>(bih, bhh, biasc);
    build_rowstart<<<(NNODES + T - 1) / T, T, 0, stream>>>(batch, rs);
    init_state   <<<(B_GRAPHS * KCOMB + T - 1) / T, T, 0, stream>>>(inbf, h, c);

    // GEMM grids: one wave per 32x32 tile, 8 waves/block
    const int zWaves    = (B_GRAPHS / 32) * (NGATES / 32);   // 2048
    const int postWaves = (B_GRAPHS / 32) * (DOUT_ / 32);    // 512

    for (int t = 0; t < 6; ++t) {
        // z = [q|r|h](bf16) @ [Wih|Whh]^T + (bih+bhh)
        gemm_bf16_wmma<<<zWaves / 8, T, 0, stream>>>(inbf, KCOMB, Wcomb, KCOMB, biasc,
                                                     z, NGATES,
                                                     B_GRAPHS, NGATES, KCOMB, 0);
        lstm_gate<<<(B_GRAPHS * DIN + T - 1) / T, T, 0, stream>>>(z, c, h, inbf);
        attn_e   <<<(NNODES * 32 + T - 1) / T, T, 0, stream>>>(x, batch, h, e);
        seg_reduce<<<B_GRAPHS, T, 0, stream>>>(e, rs, emax, denom);
        attn_coef <<<(NNODES + T - 1) / T, T, 0, stream>>>(e, batch, emax, denom, ac);
        seg_wsum  <<<B_GRAPHS, DIN / 2, 0, stream>>>(x, ac, rs, inbf);
    }

    // out = relu(q_star @ Wpost^T + bpost); q_star = inbf cols [0,1024)
    gemm_bf16_wmma<<<postWaves / 8, T, 0, stream>>>(inbf, KCOMB, Wp, 2 * DIN, bpost,
                                                    out, DOUT_,
                                                    B_GRAPHS, DOUT_, 2 * DIN, 1);
}
